// LocMatcherPN_64845416235457
// MI455X (gfx1250) — compile-verified, matrix operand order checked by reading, not run on hardware
//
#include <hip/hip_runtime.h>
#include <hip/hip_bf16.h>

typedef __attribute__((ext_vector_type(16))) _Float16 v16h;
typedef __attribute__((ext_vector_type(8)))  _Float16 v8h;
typedef __attribute__((ext_vector_type(8)))  float    v8f;

#define Bsz   128
#define Tsz   1024
#define Hsz   256
#define G4H   1024   // 4*H
#define INF_K 19     // LOC_D + TD_E

// ---- WMMA fragment index helpers (16-bit A layout per CDNA5 ISA 7.12.2;
//      B assumed symmetric with M<->N; C/D: row m=v+8*(lane>>4), col n=lane&15)
__device__ __forceinline__ int frag_lane(int mn, int kk) {
  return (mn & 15) | (((kk >> 3) & 1) << 4);
}
__device__ __forceinline__ int frag_j(int kk) {
  return (kk & 7) | ((kk >> 4) << 3);
}
__device__ __forceinline__ int frag_kk(int lane, int j) {
  return (j & 7) | (((lane >> 4) & 1) << 3) | ((j >> 3) << 4);
}
__device__ __forceinline__ float sigmoidf_(float x) { return 1.f / (1.f + __expf(-x)); }

// ======================= prep kernels (f32 -> f16 fragment swizzles) ==========
// Whhb: [dir][ntile(64)*8+ktile][lane(32)][j(16)],  B[kk][n] = Whh[n][kk]
__global__ void prep_whh_kernel(const float* __restrict__ whh_f,
                                const float* __restrict__ whh_b,
                                _Float16* __restrict__ out) {
  int idx = blockIdx.x * 256 + threadIdx.x;          // 2*64*8*32*16 = 524288
  int j = idx & 15, lane = (idx >> 4) & 31, kt = (idx >> 9) & 7;
  int nt = (idx >> 12) & 63, dir = idx >> 18;
  int n = nt * 16 + (lane & 15);
  int kk = kt * 32 + frag_kk(lane, j);
  const float* src = dir ? whh_b : whh_f;
  out[idx] = (_Float16)src[n * Hsz + kk];
}

// Wihb: [dir][ntile(64)][lane][j], K padded 19 -> 32 with zeros
__global__ void prep_wih_kernel(const float* __restrict__ wih_f,
                                const float* __restrict__ wih_b,
                                _Float16* __restrict__ out) {
  int idx = blockIdx.x * 256 + threadIdx.x;          // 2*64*32*16 = 65536
  int j = idx & 15, lane = (idx >> 4) & 31, nt = (idx >> 9) & 63, dir = idx >> 15;
  int n = nt * 16 + (lane & 15);
  int kk = frag_kk(lane, j);
  const float* src = dir ? wih_b : wih_f;
  out[idx] = (_Float16)((kk < INF_K) ? src[n * INF_K + kk] : 0.f);
}

// outWb: [ntile(2)][ktile(16)][lane][j],  B[kk][n] = out_W[n][kk]
__global__ void prep_outw_kernel(const float* __restrict__ outW,
                                 _Float16* __restrict__ out) {
  int idx = blockIdx.x * 256 + threadIdx.x;          // 2*16*32*16 = 16384
  int j = idx & 15, lane = (idx >> 4) & 31, kt = (idx >> 9) & 15, nt = idx >> 13;
  int n = nt * 16 + (lane & 15);
  int kk = kt * 32 + frag_kk(lane, j);
  out[idx] = (_Float16)outW[n * 512 + kk];
}

// addr_feat[b][o] = addr_b[o] + addr_W[o][0]*addr[b] + sum_e addr_W[o][1+e]*poi[type[b]][e]
__global__ void prep_af_kernel(const float* __restrict__ addr,
                               const int* __restrict__ addr_type,
                               const float* __restrict__ poi,
                               const float* __restrict__ aW,
                               const float* __restrict__ ab,
                               float* __restrict__ af) {
  int idx = blockIdx.x * 256 + threadIdx.x;          // 128*32 = 4096
  int b = idx >> 5, o = idx & 31;
  int at = addr_type[b];
  float v = ab[o] + aW[o * 4 + 0] * addr[b];
  #pragma unroll
  for (int e = 0; e < 3; ++e) v += aW[o * 4 + 1 + e] * poi[at * 3 + e];
  af[idx] = v;
}

// ======================= bidirectional LSTM scan ==============================
// grid = 16 blocks: blockIdx.x = bt*2 + dir. 256 threads = 8 waves.
// Wave w owns gate-columns [32w, 32w+32) of i,f,g,o  (ntile = g*16 + 2w + lt).
// h kept in LDS in A-fragment layout (wave w owns h k_tile == w). c in VGPRs.
// Weight fragments stream from L2 every step (working set 576KB << 192MB L2);
// an asm memory barrier per step prevents LICM from hoisting them into VGPRs
// (which previously caused scratch spill/reload in the hot loop).
__global__ __launch_bounds__(256, 2) void lstm_scan_kernel(
    const float* __restrict__ loc,    // (B,T,16)
    const float* __restrict__ tds,    // (B,T,8)
    const int*   __restrict__ dlen,   // (B,)
    const float* __restrict__ timeW,  // (3,8)
    const float* __restrict__ timeb,  // (3,)
    const _Float16* __restrict__ Wihb,  // [2][64][32][16]
    const _Float16* __restrict__ Whhb,  // [2][64*8][32][16]
    const float* __restrict__ bias_f, const float* __restrict__ bias_b,
    _Float16* __restrict__ gru)       // (B,T,512)  [0:256]=fwd [256:512]=bwd
{
  __shared__ _Float16 hfrag[8 * 32 * 16];   // 8KB: h, fragment layout
  __shared__ _Float16 xfrag[32 * 16];       // 1KB: x_t tile (K padded to 32)
  __shared__ int   dl_s[16];
  __shared__ float tW[24], tB[3];

  const int tid  = threadIdx.x;
  const int lane = tid & 31;
  const int w    = tid >> 5;
  const int dir  = blockIdx.x & 1;
  const int bt   = blockIdx.x >> 1;
  const int b0   = bt * 16;

  const _Float16* wih = Wihb + (size_t)dir * 64 * 512;
  const _Float16* whh = Whhb + (size_t)dir * 64 * 8 * 512;
  const float*   bias = dir ? bias_b : bias_f;

  for (int i = tid; i < 8 * 32 * 16; i += 256) hfrag[i] = (_Float16)0.f;
  if (tid < 16) dl_s[tid] = dlen[b0 + tid];
  if (tid < 24) tW[tid] = timeW[tid];
  if (tid < 3)  tB[tid] = timeb[tid];

  // per-wave bias (t-invariant): lt tile col -> gate column hk = 32w + coln
  float bi[2], bf_[2], bg[2], bo[2];
  #pragma unroll
  for (int lt = 0; lt < 2; ++lt) {
    int hk = 32 * w + (lane & 15) + 16 * lt;
    bi[lt]  = bias[0 * Hsz + hk];
    bf_[lt] = bias[1 * Hsz + hk];
    bg[lt]  = bias[2 * Hsz + hk];
    bo[lt]  = bias[3 * Hsz + hk];
  }

  float c[2][8];
  #pragma unroll
  for (int lt = 0; lt < 2; ++lt)
    #pragma unroll
    for (int v = 0; v < 8; ++v) c[lt][v] = 0.f;

  __syncthreads();

  for (int s = 0; s < Tsz; ++s) {
    // full compiler memory barrier: keep weight-fragment loads inside the loop
    // (stream from L2) instead of hoisting ~576 VGPRs of weights + spilling.
    asm volatile("" ::: "memory");

    const int tt = dir ? (Tsz - 1 - s) : s;

    // ---- A: build x_t fragment (time-proj fused), threads 0..15 do one row
    if (tid < 16) {
      int m = tid;
      const float* lrow = loc + ((size_t)(b0 + m) * Tsz + tt) * 16;
      const float* trow = tds + ((size_t)(b0 + m) * Tsz + tt) * 8;
      // prefetch next step's rows (global_prefetch_b8) to hide HBM latency
      int tn = dir ? (tt - 1) : (tt + 1);
      if (tn >= 0 && tn < Tsz) {
        __builtin_prefetch(loc + ((size_t)(b0 + m) * Tsz + tn) * 16, 0, 3);
        __builtin_prefetch(tds + ((size_t)(b0 + m) * Tsz + tn) * 8, 0, 3);
      }
      float td[3];
      #pragma unroll
      for (int e = 0; e < 3; ++e) {
        float a = tB[e];
        #pragma unroll
        for (int q = 0; q < 8; ++q) a += tW[e * 8 + q] * trow[q];
        td[e] = a;
      }
      for (int kk = 0; kk < 32; ++kk) {
        float v = (kk < 16) ? lrow[kk] : ((kk < INF_K) ? td[kk - 16] : 0.f);
        xfrag[frag_lane(m, kk) * 16 + frag_j(kk)] = (_Float16)v;
      }
    }
    __syncthreads();

    // ---- B: gates = x@Wih.T + h@Whh.T  (WMMA f32 <- f16)
    v8f acc[8];
    #pragma unroll
    for (int i = 0; i < 8; ++i) acc[i] = (v8f){};

    {
      v16h a = *(const v16h*)(xfrag + lane * 16);
      #pragma unroll
      for (int g = 0; g < 4; ++g) {
        #pragma unroll
        for (int lt = 0; lt < 2; ++lt) {
          int ntile = g * 16 + 2 * w + lt;
          v16h bm = *(const v16h*)(wih + ((size_t)ntile * 32 + lane) * 16);
          acc[g * 2 + lt] = __builtin_amdgcn_wmma_f32_16x16x32_f16(
              false, a, false, bm, (short)0, acc[g * 2 + lt], false, false);
        }
      }
    }
    #pragma unroll
    for (int kt = 0; kt < 8; ++kt) {
      v16h a = *(const v16h*)(hfrag + (kt * 32 + lane) * 16);
      #pragma unroll
      for (int g = 0; g < 4; ++g) {
        #pragma unroll
        for (int lt = 0; lt < 2; ++lt) {
          int ntile = g * 16 + 2 * w + lt;
          v16h bm = *(const v16h*)(whh + (((size_t)ntile * 8 + kt) * 32 + lane) * 16);
          acc[g * 2 + lt] = __builtin_amdgcn_wmma_f32_16x16x32_f16(
              false, a, false, bm, (short)0, acc[g * 2 + lt], false, false);
        }
      }
    }

    // ---- activations + masked c/h update (wave-local columns)
    const int mlo = (lane >> 4) << 3;
    _Float16 hout[2][8];
    #pragma unroll
    for (int lt = 0; lt < 2; ++lt) {
      int coln = (lane & 15) + 16 * lt;      // local col in [0,32)
      #pragma unroll
      for (int v = 0; v < 8; ++v) {
        int m = v + mlo;
        float gi = acc[0 * 2 + lt][v] + bi[lt];
        float gf = acc[1 * 2 + lt][v] + bf_[lt];
        float gg = acc[2 * 2 + lt][v] + bg[lt];
        float go = acc[3 * 2 + lt][v] + bo[lt];
        float cn = sigmoidf_(gf) * c[lt][v] + sigmoidf_(gi) * tanhf(gg);
        float hn = sigmoidf_(go) * tanhf(cn);
        bool upd = tt < dl_s[m];
        float hprev = (float)hfrag[(w * 32 + frag_lane(m, coln)) * 16 + frag_j(coln)];
        c[lt][v]    = upd ? cn : c[lt][v];
        hout[lt][v] = (_Float16)(upd ? hn : hprev);
      }
    }
    __syncthreads();   // all reads of old h / xfrag done

    // ---- C: publish new h into fragment LDS (wave w owns k_tile w)
    #pragma unroll
    for (int lt = 0; lt < 2; ++lt) {
      int coln = (lane & 15) + 16 * lt;
      #pragma unroll
      for (int v = 0; v < 8; ++v) {
        int m = v + mlo;
        hfrag[(w * 32 + frag_lane(m, coln)) * 16 + frag_j(coln)] = hout[lt][v];
      }
    }
    __syncthreads();

    // ---- D: write h (carried) to gru_out, 32B per thread
    {
      int m  = tid & 15;
      int kc = (tid >> 4) * 16;
      v16h tv;
      #pragma unroll
      for (int q = 0; q < 16; ++q) {
        int k = kc + q;
        tv[q] = hfrag[((k >> 5) * 32 + frag_lane(m, k & 31)) * 16 + frag_j(k & 31)];
      }
      _Float16* dst = gru + ((size_t)(b0 + m) * Tsz + tt) * 512 + dir * Hsz + kc;
      *(v16h*)dst = tv;
    }
  }
}

// ======================= output projection + score ============================
// hidden = tanh(gru@out_W.T + addr_feat);  scores = hidden @ comb_W.T
// one M-tile (16 rows of B*T) per wave; N=32 (2 tiles), K=512 (16 tiles)
__global__ __launch_bounds__(256, 2) void outproj_kernel(
    const _Float16* __restrict__ gru,     // (B*T, 512) f16
    const _Float16* __restrict__ outWb,   // [2][16][32][16]
    const float* __restrict__ af,         // (B,32)
    const float* __restrict__ combW,      // (32,)
    float* __restrict__ scores)           // (B*T,)
{
  __shared__ _Float16 wlds[2 * 16 * 32 * 16];  // 32KB
  __shared__ float cw[32];
  const int tid = threadIdx.x, lane = tid & 31, w = tid >> 5;

  {
    const int* src = (const int*)outWb;
    int* dst = (int*)wlds;
    for (int i = tid; i < 8192; i += 256) dst[i] = src[i];
  }
  if (tid < 32) cw[tid] = combW[tid];
  __syncthreads();

  const int mt = blockIdx.x * 8 + w;       // 0..8191
  const int r0 = mt * 16;
  const int b  = mt >> 6;                  // 64 tiles per batch row

  v8f acc0 = (v8f){}, acc1 = (v8f){};
  const int r     = r0 + (lane & 15);
  const int kskew = (lane >> 4) << 3;
  #pragma unroll
  for (int kt = 0; kt < 16; ++kt) {
    const _Float16* p = gru + (size_t)r * 512 + kt * 32 + kskew;
    v8h alo = *(const v8h*)p;
    v8h ahi = *(const v8h*)(p + 16);
    v16h a;
    #pragma unroll
    for (int q = 0; q < 8; ++q) { a[q] = alo[q]; a[q + 8] = ahi[q]; }
    v16h b0 = *(const v16h*)(wlds + ((0 * 16 + kt) * 32 + lane) * 16);
    v16h b1 = *(const v16h*)(wlds + ((1 * 16 + kt) * 32 + lane) * 16);
    acc0 = __builtin_amdgcn_wmma_f32_16x16x32_f16(false, a, false, b0, (short)0, acc0, false, false);
    acc1 = __builtin_amdgcn_wmma_f32_16x16x32_f16(false, a, false, b1, (short)0, acc1, false, false);
  }

  const int n   = lane & 15;
  const int mlo = (lane >> 4) << 3;
  const float af0 = af[b * 32 + n],      af1 = af[b * 32 + 16 + n];
  const float cw0 = cw[n],               cw1 = cw[16 + n];
  #pragma unroll
  for (int v = 0; v < 8; ++v) {
    float h0 = tanhf(acc0[v] + af0);
    float h1 = tanhf(acc1[v] + af1);
    float val = h0 * cw0 + h1 * cw1;
    val += __shfl_xor(val, 1);
    val += __shfl_xor(val, 2);
    val += __shfl_xor(val, 4);
    val += __shfl_xor(val, 8);            // row-sums within each 16-lane half
    if ((lane & 15) == 0) scores[r0 + v + mlo] = val;
  }
}

// ======================= masked log-softmax over T ============================
__global__ __launch_bounds__(256, 1) void softmax_kernel(
    const float* __restrict__ scores, const int* __restrict__ dlen,
    float* __restrict__ out)
{
  __shared__ float red[256];
  const int b = blockIdx.x, tid = threadIdx.x;
  const int dl = dlen[b];
  float sv[4];
  float mx = -3.4e38f;
  #pragma unroll
  for (int i = 0; i < 4; ++i) {
    int t = tid + i * 256;
    float s = scores[b * Tsz + t];
    s = (t < dl) ? s : -1e30f;
    sv[i] = s;
    mx = fmaxf(mx, s);
  }
  red[tid] = mx; __syncthreads();
  for (int off = 128; off > 0; off >>= 1) {
    if (tid < off) red[tid] = fmaxf(red[tid], red[tid + off]);
    __syncthreads();
  }
  mx = red[0]; __syncthreads();
  float sum = 0.f;
  #pragma unroll
  for (int i = 0; i < 4; ++i) sum += __expf(sv[i] - mx);
  red[tid] = sum; __syncthreads();
  for (int off = 128; off > 0; off >>= 1) {
    if (tid < off) red[tid] += red[tid + off];
    __syncthreads();
  }
  const float lse = mx + __logf(red[0]);
  #pragma unroll
  for (int i = 0; i < 4; ++i) {
    int t = tid + i * 256;
    out[b * Tsz + t] = (t < dl) ? (sv[i] - lse) : 0.f;
  }
}

// ======================= launch ==============================================
extern "C" void kernel_launch(void* const* d_in, const int* in_sizes, int n_in,
                              void* d_out, int out_size, void* d_ws, size_t ws_size,
                              hipStream_t stream) {
  const float* addr      = (const float*)d_in[0];
  const int*   addr_type = (const int*)  d_in[1];
  const float* loc       = (const float*)d_in[2];
  const float* tds       = (const float*)d_in[3];
  const int*   dlen      = (const int*)  d_in[4];
  const float* poi       = (const float*)d_in[5];
  const float* timeW     = (const float*)d_in[6];
  const float* timeb     = (const float*)d_in[7];
  const float* addrW     = (const float*)d_in[8];
  const float* addrb     = (const float*)d_in[9];
  const float* wih_f     = (const float*)d_in[10];
  const float* whh_f     = (const float*)d_in[11];
  const float* b_f       = (const float*)d_in[12];
  const float* wih_b     = (const float*)d_in[13];
  const float* whh_b     = (const float*)d_in[14];
  const float* b_b       = (const float*)d_in[15];
  const float* outW      = (const float*)d_in[16];
  const float* combW     = (const float*)d_in[17];
  float* out = (float*)d_out;

  // workspace carve-up (256B aligned):
  // Whhb 1MB | Wihb 128KB | outWb 32KB | addr_feat 16KB | gru 128MB | scores 512KB
  char* ws = (char*)d_ws;
  size_t off = 0;
  auto take = [&](size_t bytes) {
    char* p = ws + off;
    off += (bytes + 255) & ~(size_t)255;
    return p;
  };
  _Float16* Whhb   = (_Float16*)take((size_t)2 * 64 * 8 * 32 * 16 * 2);
  _Float16* Wihb   = (_Float16*)take((size_t)2 * 64 * 32 * 16 * 2);
  _Float16* outWb  = (_Float16*)take((size_t)2 * 16 * 32 * 16 * 2);
  float*    afeat  = (float*)   take((size_t)Bsz * 32 * 4);
  _Float16* gru    = (_Float16*)take((size_t)Bsz * Tsz * 512 * 2);
  float*    scores = (float*)   take((size_t)Bsz * Tsz * 4);
  (void)ws_size; (void)n_in; (void)in_sizes; (void)out_size;

  prep_whh_kernel <<<2048, 256, 0, stream>>>(whh_f, whh_b, Whhb);
  prep_wih_kernel <<<256,  256, 0, stream>>>(wih_f, wih_b, Wihb);
  prep_outw_kernel<<<64,   256, 0, stream>>>(outW, outWb);
  prep_af_kernel  <<<16,   256, 0, stream>>>(addr, addr_type, poi, addrW, addrb, afeat);

  lstm_scan_kernel<<<16, 256, 0, stream>>>(loc, tds, dlen, timeW, timeb,
                                           Wihb, Whhb, b_f, b_b, gru);

  outproj_kernel<<<1024, 256, 0, stream>>>(gru, outWb, afeat, combW, scores);
  softmax_kernel<<<Bsz, 256, 0, stream>>>(scores, dlen, out);
}
